// ButterflyLayer1D_35459249995889
// MI455X (gfx1250) — matrix operand order repaired and checked by reading, not compiled, not run on hardware
//
#include <hip/hip_runtime.h>

typedef __attribute__((ext_vector_type(16))) __bf16        v16bf;
typedef __attribute__((ext_vector_type(2)))  __bf16        v2bf;
typedef __attribute__((ext_vector_type(8)))  float         v8f;
typedef __attribute__((ext_vector_type(8)))  unsigned int  v8u;

#define NS     8                 // samples per workgroup
#define PITCH  132               // ushorts per position row (128 + pad)
#define SSTR   (64 * PITCH)      // ushorts per sample  (8448)
#define BUFU   (NS * SSTR)       // ushorts per state buffer (67584)
#define SMEM_BYTES (2 * BUFU * 2)// 270336 bytes (<320KB WGP LDS)

__device__ __forceinline__ unsigned short f2bf(float f) {
  return __builtin_bit_cast(unsigned short, (__bf16)f);   // hw cvt, RTNE
}
__device__ __forceinline__ unsigned pack2bf(float lo, float hi) {
  v2bf t = {(__bf16)lo, (__bf16)hi};                      // v_cvt_pk_bf16_f32
  return __builtin_bit_cast(unsigned, t);
}

// ---------------- weight pre-conversion: f32 [G][K2][128][128] -> bf16 Wt[G][128][K2*128]
// Wt[g][d][k*128+c] = src[g][k][c][d]   (transposed so B fragments are contiguous dword pairs)
__global__ void convert_w(const float* __restrict__ src, unsigned short* __restrict__ dst,
                          int K2, int total) {
  int tid = blockIdx.x * blockDim.x + threadIdx.x;
  if (tid >= total) return;
  int per = K2 << 14;                 // K2*16384
  int g   = tid / per;
  int r   = tid - g * per;
  int k   = r >> 14;
  int c   = (r >> 7) & 127;
  int d   = r & 127;
  dst[(size_t)(g * 128 + d) * (size_t)(K2 << 7) + (k << 7) + c] =
      f2bf(src[(size_t)((g * K2 + k) * 128 + c) * 128 + d]);
}

// ---------------- fused butterfly network ----------------
struct Params {
  const float* x;            // (1024, 8192) f32
  const float* bias[8];      // xb,b1,b2,b3,mb,b4,b5,b6 (f32)
  const unsigned short* w;   // bf16 transposed weights in ws
  float* out;                // (1024, 8192) f32
};

// MODE: 0 = input proj (A from global), 1 = down level, 2 = mid, 3 = up level, 4 = final proj
// flat row r -> sample s, input positions p0/p1, weight key, output position pout
template <int MODE, int LGL>
__device__ __forceinline__ void rowmap(int r, int& s, int& p0, int& p1, int& key, int& pout) {
  if constexpr (MODE == 0 || MODE == 4) {
    s = r >> 6; p0 = r & 63; p1 = p0; key = 0; pout = p0;
  } else if constexpr (MODE == 1) {            // down: pairs (2l,2l+1) of parent block
    constexpr int L = 1 << LGL;
    key     = r >> (3 + LGL);
    int rem = r & ((8 << LGL) - 1);
    s       = rem >> LGL;
    int l   = rem & (L - 1);
    pout    = (key << LGL) + l;
    p0      = ((key >> 1) << (LGL + 1)) + (l << 1);
    p1      = p0 + 1;
  } else if constexpr (MODE == 2) {            // mid: block/pos transpose, per-(k,x) weight
    int t = r >> 4, M = r & 15;
    s = M & 7;
    int k = t >> 3, xx = t & 7;
    p0 = k * 8 + xx; p1 = p0;
    pout = xx * 8 + k;
    key = t;
  } else {                                     // up: concat blocks (2x,2x+1) at pos l
    constexpr int L = 1 << LGL;
    key     = r >> (3 + LGL);
    int rem = r & ((8 << LGL) - 1);
    s       = rem >> LGL;
    int l   = rem & (L - 1);
    int xx  = key >> 1, j = key & 1;
    p0      = ((xx << 1) << LGL) + l;
    p1      = p0 + L;
    pout    = (xx << (LGL + 1)) + (l << 1) + j;
  }
}

template <int MODE, int LGL, int KSTEPS, int MT>
__device__ __forceinline__ void run_layer(unsigned short* smem, unsigned curOff, unsigned nxtOff,
                                          const float* __restrict__ x,
                                          const float* __restrict__ bias,
                                          const unsigned short* __restrict__ wbase,
                                          float* __restrict__ out, int n0) {
  constexpr int K = KSTEPS * 32;
  const int lane  = threadIdx.x & 31;
  const int wave  = threadIdx.x >> 5;
  const int h     = lane >> 4;
  const int nn    = lane & 15;

  for (int tile0 = wave; tile0 < MT; tile0 += 8) {
    const int tile = __builtin_amdgcn_readfirstlane(tile0);   // wave-uniform -> SGPR

    // per-lane A-row mapping (A rows: M = lane&15 in both halves)
    int s, p0, p1, keyL, pd;
    rowmap<MODE, LGL>(tile * 16 + nn, s, p0, p1, keyL, pd);
    // uniform weight key from the tile base row
    int su, q0, q1, keyU, pu;
    rowmap<MODE, LGL>(tile * 16, su, q0, q1, keyU, pu);
    (void)su; (void)q0; (void)q1; (void)pu; (void)keyL; (void)pd;

    const unsigned short* wlay = wbase + (unsigned)keyU * (unsigned)(128 * K); // SGPR base
    const unsigned ab0 = curOff + (unsigned)(s * SSTR + p0 * PITCH);
    const unsigned ab1 = curOff + (unsigned)(s * SSTR + p1 * PITCH);
    const float* xrow = nullptr;
    if constexpr (MODE == 0) xrow = x + (((size_t)(n0 + s)) << 13) + ((size_t)p0 << 7);

    v8f acc[8] = {};

    // one 16x16x32 K-step: ktg = global k-tile (weights), ktl = k-tile within input row
    auto kstep = [&](int ktg, int ktl, unsigned ab) {
      // ---- A fragment (16x32 bf16): VGPR v holds K = {2v,2v+1}+8h (v<4) / {16+2(v-4),..}+8h
      v8u au;
      #pragma unroll
      for (int v = 0; v < 8; ++v) {
        const int kl = ((v < 4) ? (v << 1) : (8 + (v << 1))) + (h << 3);
        if constexpr (MODE == 0) {
          const int kk = ktg * 32 + kl;
          au[v] = pack2bf(xrow[kk], xrow[kk + 1]);
        } else {
          au[v] = *(const unsigned int*)&smem[ab + (unsigned)(ktl * 32 + kl)];
        }
      }
      v16bf A = __builtin_bit_cast(v16bf, au);

      // ---- B fragments (32x16 bf16): lane -> N = nn, VGPR v holds K = {2v,2v+1}+8h (v<4) / {16+2(v-4),..}+8h
      #pragma unroll
      for (int nt = 0; nt < 8; ++nt) {
        const unsigned wo = (unsigned)nn * K + (unsigned)(nt * 16 * K + ktg * 32 + (h << 3));
        v8u bu;
        #pragma unroll
        for (int v = 0; v < 8; ++v)
          bu[v] = *(const unsigned int*)(wlay + wo + ((v < 4) ? (v << 1) : (8 + (v << 1))));
        v16bf B = __builtin_bit_cast(v16bf, bu);
        acc[nt] = __builtin_amdgcn_wmma_f32_16x16x32_bf16(
            false, A, false, B, (short)0, acc[nt], false, false);
      }
    };

    if constexpr (KSTEPS == 4) {          // K=128: single input row
      #pragma unroll 2
      for (int kt = 0; kt < 4; ++kt) kstep(kt, kt, ab0);
    } else {                              // K=256: first row then second row
      #pragma unroll 2
      for (int kt = 0; kt < 4; ++kt) kstep(kt, kt, ab0);
      #pragma unroll 2
      for (int kt = 0; kt < 4; ++kt) kstep(kt + 4, kt, ab1);
    }

    // ---- epilogue: C/D layout: VGPR v -> M = v + 8h, N = nn
    int s2[8], po2[8];
    #pragma unroll
    for (int v = 0; v < 8; ++v) {
      int d0, d1, k2;
      rowmap<MODE, LGL>(tile * 16 + v + (h << 3), s2[v], d0, d1, k2, po2[v]);
      (void)d0; (void)d1; (void)k2;
    }
    const bool valid = (MODE != 2) || (h == 0);   // mid: only 8 valid rows per tile
    #pragma unroll
    for (int nt = 0; nt < 8; ++nt) {
      const int col = nt * 16 + nn;
      float b = 0.f;
      if constexpr (MODE != 4) b = bias[keyU * 128 + col];
      #pragma unroll
      for (int v = 0; v < 8; ++v) {
        float val = acc[nt][v] + b;
        if constexpr (MODE == 4) {
          out[(((size_t)(n0 + s2[v])) << 13) + ((size_t)po2[v] << 7) + col] = val;
        } else {
          val = fmaxf(val, 0.f);
          if (valid) smem[nxtOff + (unsigned)(s2[v] * SSTR + po2[v] * PITCH + col)] = f2bf(val);
        }
      }
    }
  }
}

__global__ __launch_bounds__(256, 1) void butterfly_fused(Params pr) {
  extern __shared__ unsigned short smem[];
  const int n0 = blockIdx.x * NS;
  constexpr unsigned B0 = 0, B1 = BUFU;

  run_layer<0, 6, 4, 32>(smem, B0, B1, pr.x, pr.bias[0], pr.w + 0,       nullptr, n0); // xf
  __syncthreads();
  run_layer<1, 5, 8, 32>(smem, B1, B0, nullptr, pr.bias[1], pr.w + 16384,   nullptr, n0); // f1
  __syncthreads();
  run_layer<1, 4, 8, 32>(smem, B0, B1, nullptr, pr.bias[2], pr.w + 81920,   nullptr, n0); // f2
  __syncthreads();
  run_layer<1, 3, 8, 32>(smem, B1, B0, nullptr, pr.bias[3], pr.w + 212992,  nullptr, n0); // f3
  __syncthreads();
  run_layer<2, 3, 4, 64>(smem, B0, B1, nullptr, pr.bias[4], pr.w + 475136,  nullptr, n0); // md
  __syncthreads();
  run_layer<3, 3, 8, 32>(smem, B1, B0, nullptr, pr.bias[5], pr.w + 1523712, nullptr, n0); // f4
  __syncthreads();
  run_layer<3, 4, 8, 32>(smem, B0, B1, nullptr, pr.bias[6], pr.w + 1785856, nullptr, n0); // f5
  __syncthreads();
  run_layer<3, 5, 8, 32>(smem, B1, B0, nullptr, pr.bias[7], pr.w + 1916928, nullptr, n0); // f6
  __syncthreads();
  run_layer<4, 6, 4, 32>(smem, B0, B1, nullptr, nullptr,   pr.w + 1982464, pr.out,  n0); // kf
}

extern "C" void kernel_launch(void* const* d_in, const int* in_sizes, int n_in,
                              void* d_out, int out_size, void* d_ws, size_t ws_size,
                              hipStream_t stream) {
  (void)in_sizes; (void)n_in; (void)out_size; (void)ws_size;
  unsigned short* w = (unsigned short*)d_ws;

  // d_in order: x, xf, xb, f1, b1, f2, b2, f3, b3, md, mb, f4, b4, f5, b5, f6, b6, kf
  static const int src_idx[9] = {1, 3, 5, 7, 9, 11, 13, 15, 17};
  static const int offs[9]    = {0, 16384, 81920, 212992, 475136,
                                 1523712, 1785856, 1916928, 1982464};
  static const int Gs[9]      = {1, 2, 4, 8, 64, 8, 4, 2, 1};
  static const int K2s[9]     = {1, 2, 2, 2, 1, 2, 2, 2, 1};
  for (int i = 0; i < 9; ++i) {
    int total = Gs[i] * K2s[i] * 16384;
    convert_w<<<(total + 255) / 256, 256, 0, stream>>>(
        (const float*)d_in[src_idx[i]], w + offs[i], K2s[i], total);
  }

  Params pr;
  pr.x = (const float*)d_in[0];
  static const int bidx[8] = {2, 4, 6, 8, 10, 12, 14, 16};
  for (int i = 0; i < 8; ++i) pr.bias[i] = (const float*)d_in[bidx[i]];
  pr.w = w;
  pr.out = (float*)d_out;

  hipFuncSetAttribute((const void*)butterfly_fused,
                      hipFuncAttributeMaxDynamicSharedMemorySize, SMEM_BYTES);
  butterfly_fused<<<dim3(1024 / NS), dim3(256), SMEM_BYTES, stream>>>(pr);
}
